// MeshGraphNet_90709709291679
// MI455X (gfx1250) — matrix operand order repaired and checked by reading, not compiled, not run on hardware
//
#include <hip/hip_runtime.h>

#define HID     128
#define NLAYERS 12
#define NNODES  50000
#define NEDGES  400000
#define IN_NODE 26
#define IN_EDGE 3

typedef __attribute__((ext_vector_type(16))) __bf16 v16bf;
typedef __attribute__((ext_vector_type(8)))  float  v8f;

// Native f32 atomic add (GLOBAL_ATOMIC_ADD_F32) instead of a CAS loop.
// agg/deg are private scratch accumulators, so unsafe HW atomics are fine.
__device__ __forceinline__ void atomic_add_f32(float* p, float v) {
#if defined(__gfx1250__)
    unsafeAtomicAdd(p, v);
#else
    atomicAdd(p, v);
#endif
}

// ---------------------------------------------------------------------------
// Fragment helpers (CDNA5 WMMA 16x16x32 bf16 layouts, wave32)
// ---------------------------------------------------------------------------

// A-matrix 16x32 bf16 tile: lane 0-15 -> row M=lane, K halves {kHalf..kHalf+7,
// 16+kHalf..16+kHalf+7} with kHalf = 0 (lanes 0-15) or 8 (lanes 16-31).
// Rows are contiguous in memory -> two 16B loads per lane.
__device__ __forceinline__ v16bf load_afrag_bf(const __bf16* rowBase, int kt, int lane) {
    const int kHalf = (lane & 16) >> 1;                // 0 or 8
    const __bf16* p = rowBase + kt * 32 + kHalf;
    union { uint4 u[2]; v16bf v; } u;
    u.u[0] = *(const uint4*)(p);                       // K = kHalf .. kHalf+7
    u.u[1] = *(const uint4*)(p + 16);                  // K = 16+kHalf .. +7
    return u.v;
}

// Same A layout but sourced from an f32 row with a per-row scale (agg * invdeg).
__device__ __forceinline__ v16bf load_afrag_f32(const float* rowBase, int kt,
                                                int lane, float scale) {
    const int kHalf = (lane & 16) >> 1;
    v16bf r;
#pragma unroll
    for (int h = 0; h < 16; ++h) {
        const int K = kt * 32 + kHalf + h + ((h < 8) ? 0 : 8);
        r[h] = (__bf16)(rowBase[K] * scale);
    }
    return r;
}

// B-matrix 32x16 tile from the pre-packed weight stream: each lane's 16 halves
// are 32 contiguous bytes (packing done by repack_kernel below).
__device__ __forceinline__ v16bf load_bfrag(const __bf16* pk, int kt, int nt, int lane) {
    return *(const v16bf*)(pk + (((size_t)kt * 8 + nt) * 32 + lane) * 16);
}

// C/D 16x16 f32 layout: element i of v8f is (M = i + 8*(lane>=16), N = lane&15).
// Store bias+ReLU result as bf16 into a 16x128 LDS tile (A-layout friendly).
__device__ __forceinline__ void store_c_relu(__bf16* sm, const v8f& c, int nt,
                                             int lane, float bias) {
    const int Mb = (lane & 16) >> 1;
    const int n  = nt * 16 + (lane & 15);
#pragma unroll
    for (int i = 0; i < 8; ++i) {
        float v = fmaxf(c[i] + bias, 0.0f);
        sm[(Mb + i) * HID + n] = (__bf16)v;
    }
}

// ---------------------------------------------------------------------------
// Weight repack: f32 (K x 128) row-major -> packed bf16 WMMA B tiles.
// B-tile layout matching A: lanes 0-15 hold K={0..7,16..23},
// lanes 16-31 hold K={8..15,24..31} (half h -> K=kHalf+h+(h<8?0:8)).
// ---------------------------------------------------------------------------
__global__ void repack_kernel(const float* __restrict__ W, __bf16* __restrict__ P,
                              int Ksrc, int Ktot, int layers) {
    const size_t perLayer = (size_t)(Ktot / 32) * 8 * 512;
    const size_t total    = perLayer * layers;
    const size_t idx = (size_t)blockIdx.x * 256 + threadIdx.x;
    if (idx >= total) return;
    const int    l    = (int)(idx / perLayer);
    const size_t r    = idx % perLayer;
    const int    h    = (int)(r & 15);
    const int    lane = (int)((r >> 4) & 31);
    const int    tl   = (int)(r >> 9);
    const int    nt   = tl & 7;
    const int    kt   = tl >> 3;
    const int    K    = kt * 32 + ((lane & 16) ? 8 : 0) + h + ((h < 8) ? 0 : 8);
    const int    n    = nt * 16 + (lane & 15);
    float v = (K < Ksrc) ? W[((size_t)l * Ksrc + K) * HID + n] : 0.0f;
    P[idx] = (__bf16)v;
}

__global__ void zero_kernel(float* __restrict__ agg, float* __restrict__ deg) {
    const size_t i = (size_t)blockIdx.x * 256 + threadIdx.x;
    if (i < (size_t)NNODES * HID) agg[i] = 0.0f;
    if (i < NNODES) deg[i] = 0.0f;
}

__global__ void deg_kernel(const int* __restrict__ dstI, float* __restrict__ deg) {
    const int e = blockIdx.x * 256 + threadIdx.x;
    if (e < NEDGES) atomic_add_f32(deg + dstI[e], 1.0f);
}

__global__ void invdeg_kernel(float* __restrict__ deg) {
    const int i = blockIdx.x * 256 + threadIdx.x;
    if (i < NNODES) deg[i] = 1.0f / fmaxf(deg[i], 1.0f);
}

// ---------------------------------------------------------------------------
// Encoders: padded-K (26->32, 3->32) MLP2 per 16-row tile, one wave per tile.
// ---------------------------------------------------------------------------
__global__ __launch_bounds__(32) void enc_node_kernel(
        const float* __restrict__ x, const __bf16* __restrict__ w1p,
        const float* __restrict__ b1, const __bf16* __restrict__ w2p,
        const float* __restrict__ b2, float* __restrict__ hf,
        __bf16* __restrict__ hbf) {
    __shared__ alignas(16) __bf16 sm[16 * HID];
    const int lane = threadIdx.x;
    const int tile = blockIdx.x;
    const int row  = lane & 15;
    const float* xr = x + (size_t)(tile * 16 + row) * IN_NODE;
    const int kHalf = (lane & 16) >> 1;
    v16bf a;
#pragma unroll
    for (int h = 0; h < 16; ++h) {
        const int K = kHalf + h + ((h < 8) ? 0 : 8);
        a[h] = (K < IN_NODE) ? (__bf16)xr[K] : (__bf16)0.0f;
    }
#pragma unroll
    for (int nt = 0; nt < 8; ++nt) {
        v8f c = {};
        v16bf b = load_bfrag(w1p, 0, nt, lane);
        c = __builtin_amdgcn_wmma_f32_16x16x32_bf16(false, a, false, b, (short)0, c, false, false);
        store_c_relu(sm, c, nt, lane, b1[nt * 16 + row]);
    }
    __syncthreads();
    v16bf a2[4];
#pragma unroll
    for (int kt = 0; kt < 4; ++kt) a2[kt] = load_afrag_bf(sm + row * HID, kt, lane);
#pragma unroll
    for (int nt = 0; nt < 8; ++nt) {
        v8f c = {};
#pragma unroll
        for (int kt = 0; kt < 4; ++kt) {
            v16bf b = load_bfrag(w2p, kt, nt, lane);
            c = __builtin_amdgcn_wmma_f32_16x16x32_bf16(false, a2[kt], false, b, (short)0, c, false, false);
        }
        const float bias = b2[nt * 16 + row];
        const int Mb = (lane & 16) >> 1;
        const int n  = nt * 16 + row;
#pragma unroll
        for (int i = 0; i < 8; ++i) {
            float v = fmaxf(c[i] + bias, 0.0f);
            const size_t node = (size_t)tile * 16 + Mb + i;
            hf[node * HID + n]  = v;
            hbf[node * HID + n] = (__bf16)v;
        }
    }
}

__global__ __launch_bounds__(32) void enc_edge_kernel(
        const float* __restrict__ ea, const __bf16* __restrict__ w1p,
        const float* __restrict__ b1, const __bf16* __restrict__ w2p,
        const float* __restrict__ b2, __bf16* __restrict__ ebf) {
    __shared__ alignas(16) __bf16 sm[16 * HID];
    const int lane = threadIdx.x;
    const int tile = blockIdx.x;
    const int row  = lane & 15;
    const float* xr = ea + (size_t)(tile * 16 + row) * IN_EDGE;
    const int kHalf = (lane & 16) >> 1;
    v16bf a;
#pragma unroll
    for (int h = 0; h < 16; ++h) {
        const int K = kHalf + h + ((h < 8) ? 0 : 8);
        a[h] = (K < IN_EDGE) ? (__bf16)xr[K] : (__bf16)0.0f;
    }
#pragma unroll
    for (int nt = 0; nt < 8; ++nt) {
        v8f c = {};
        v16bf b = load_bfrag(w1p, 0, nt, lane);
        c = __builtin_amdgcn_wmma_f32_16x16x32_bf16(false, a, false, b, (short)0, c, false, false);
        store_c_relu(sm, c, nt, lane, b1[nt * 16 + row]);
    }
    __syncthreads();
    v16bf a2[4];
#pragma unroll
    for (int kt = 0; kt < 4; ++kt) a2[kt] = load_afrag_bf(sm + row * HID, kt, lane);
#pragma unroll
    for (int nt = 0; nt < 8; ++nt) {
        v8f c = {};
#pragma unroll
        for (int kt = 0; kt < 4; ++kt) {
            v16bf b = load_bfrag(w2p, kt, nt, lane);
            c = __builtin_amdgcn_wmma_f32_16x16x32_bf16(false, a2[kt], false, b, (short)0, c, false, false);
        }
        const float bias = b2[nt * 16 + row];
        const int Mb = (lane & 16) >> 1;
        const int n  = nt * 16 + row;
#pragma unroll
        for (int i = 0; i < 8; ++i) {
            float v = fmaxf(c[i] + bias, 0.0f);
            ebf[(size_t)(tile * 16 + Mb + i) * HID + n] = (__bf16)v;
        }
    }
}

// ---------------------------------------------------------------------------
// Message MLP: per wave, 16 edges. Gather h[dst],h[src],e as bf16 A-fragments
// (384-wide input = 12 K-tiles), MLP2, then scatter fp32 atomics into agg.
// 4 waves per block, disjoint LDS slices.
// ---------------------------------------------------------------------------
__global__ __launch_bounds__(128) void msg_kernel(
        const __bf16* __restrict__ hbf, const __bf16* __restrict__ ebf,
        const int* __restrict__ srcI, const int* __restrict__ dstI,
        const __bf16* __restrict__ w1p, const float* __restrict__ b1,
        const __bf16* __restrict__ w2p, const float* __restrict__ b2,
        float* __restrict__ agg) {
    __shared__ alignas(16) __bf16 sm[4][16 * HID];
    __shared__ int sdst[4][16];
    const int wave = threadIdx.x >> 5;
    const int lane = threadIdx.x & 31;
    const int tile = blockIdx.x * 4 + wave;
    const int row  = lane & 15;
    const int edge = tile * 16 + row;
    const int s = srcI[edge];
    const int d = dstI[edge];
    if (lane < 16) sdst[wave][lane] = d;

    v16bf a[12];
    const __bf16* hd = hbf + (size_t)d * HID;
    const __bf16* hs = hbf + (size_t)s * HID;
    const __bf16* er = ebf + (size_t)edge * HID;
#pragma unroll
    for (int t = 0; t < 4; ++t) {
        a[t]     = load_afrag_bf(hd, t, lane);
        a[4 + t] = load_afrag_bf(hs, t, lane);
        a[8 + t] = load_afrag_bf(er, t, lane);
    }
    __bf16* smw = sm[wave];
#pragma unroll
    for (int nt = 0; nt < 8; ++nt) {
        v8f c = {};
#pragma unroll
        for (int kt = 0; kt < 12; ++kt) {
            v16bf b = load_bfrag(w1p, kt, nt, lane);
            c = __builtin_amdgcn_wmma_f32_16x16x32_bf16(false, a[kt], false, b, (short)0, c, false, false);
        }
        store_c_relu(smw, c, nt, lane, b1[nt * 16 + row]);
    }
    __syncthreads();
    v16bf a2[4];
#pragma unroll
    for (int kt = 0; kt < 4; ++kt) a2[kt] = load_afrag_bf(smw + row * HID, kt, lane);
#pragma unroll
    for (int nt = 0; nt < 8; ++nt) {
        v8f c = {};
#pragma unroll
        for (int kt = 0; kt < 4; ++kt) {
            v16bf b = load_bfrag(w2p, kt, nt, lane);
            c = __builtin_amdgcn_wmma_f32_16x16x32_bf16(false, a2[kt], false, b, (short)0, c, false, false);
        }
        const float bias = b2[nt * 16 + row];
        const int Mb = (lane & 16) >> 1;
        const int n  = nt * 16 + row;
#pragma unroll
        for (int i = 0; i < 8; ++i) {
            float v = fmaxf(c[i] + bias, 0.0f);
            const int dn = sdst[wave][Mb + i];
            atomic_add_f32(agg + (size_t)dn * HID + n, v);
        }
    }
}

// ---------------------------------------------------------------------------
// Update MLP: per wave, 16 nodes. Input = [h (bf16), agg*invdeg (f32->bf16)],
// MLP2, residual add, refresh hf/hbf, zero agg for next layer.
// ---------------------------------------------------------------------------
__global__ __launch_bounds__(32) void upd_kernel(
        float* __restrict__ hf, __bf16* __restrict__ hbf,
        float* __restrict__ agg, const float* __restrict__ invdeg,
        const __bf16* __restrict__ w1p, const float* __restrict__ b1,
        const __bf16* __restrict__ w2p, const float* __restrict__ b2) {
    __shared__ alignas(16) __bf16 sm[16 * HID];
    const int lane = threadIdx.x;
    const int tile = blockIdx.x;
    const int row  = lane & 15;
    const size_t node = (size_t)tile * 16 + row;
    const float  sc   = invdeg[node];
    v16bf a[8];
    const __bf16* hr = hbf + node * HID;
    const float*  ar = agg + node * HID;
#pragma unroll
    for (int t = 0; t < 4; ++t) {
        a[t]     = load_afrag_bf(hr, t, lane);
        a[4 + t] = load_afrag_f32(ar, t, lane, sc);
    }
#pragma unroll
    for (int nt = 0; nt < 8; ++nt) {
        v8f c = {};
#pragma unroll
        for (int kt = 0; kt < 8; ++kt) {
            v16bf b = load_bfrag(w1p, kt, nt, lane);
            c = __builtin_amdgcn_wmma_f32_16x16x32_bf16(false, a[kt], false, b, (short)0, c, false, false);
        }
        store_c_relu(sm, c, nt, lane, b1[nt * 16 + row]);
    }
    __syncthreads();
    v16bf a2[4];
#pragma unroll
    for (int kt = 0; kt < 4; ++kt) a2[kt] = load_afrag_bf(sm + row * HID, kt, lane);
#pragma unroll
    for (int nt = 0; nt < 8; ++nt) {
        v8f c = {};
#pragma unroll
        for (int kt = 0; kt < 4; ++kt) {
            v16bf b = load_bfrag(w2p, kt, nt, lane);
            c = __builtin_amdgcn_wmma_f32_16x16x32_bf16(false, a2[kt], false, b, (short)0, c, false, false);
        }
        const float bias = b2[nt * 16 + row];
        const int Mb = (lane & 16) >> 1;
        const int n  = nt * 16 + row;
#pragma unroll
        for (int i = 0; i < 8; ++i) {
            const float u  = fmaxf(c[i] + bias, 0.0f);
            const size_t nd = (size_t)tile * 16 + Mb + i;
            const float hv = hf[nd * HID + n] + u;
            hf[nd * HID + n]  = hv;
            hbf[nd * HID + n] = (__bf16)hv;
            agg[nd * HID + n] = 0.0f;   // ready for next layer
        }
    }
}

// ---------------------------------------------------------------------------
// Readout: conv1 (1->8, k=15, stride 4) + ReLU, conv2 (8->1, k=10) per node.
// One thread per (node, output position); recompute conv1 windows (cheap).
// ---------------------------------------------------------------------------
__global__ void conv_kernel(const float* __restrict__ hf,
                            const float* __restrict__ w1, const float* __restrict__ b1,
                            const float* __restrict__ w2, const float* __restrict__ b2,
                            float* __restrict__ out) {
    const size_t idx = (size_t)blockIdx.x * 256 + threadIdx.x;
    if (idx >= (size_t)NNODES * 20) return;
    const int node = (int)(idx / 20);
    const int q    = (int)(idx % 20);
    const float* r = hf + (size_t)node * HID;
    float s = b2[0];
#pragma unroll 1
    for (int k = 0; k < 10; ++k) {
        const int p = q + k;
#pragma unroll
        for (int c = 0; c < 8; ++c) {
            float acc = b1[c];
#pragma unroll
            for (int t = 0; t < 15; ++t) acc += r[4 * p + t] * w1[c * 15 + t];
            s += fmaxf(acc, 0.0f) * w2[c * 10 + k];
        }
    }
    out[idx] = s;
}

// ---------------------------------------------------------------------------
// Launch
// ---------------------------------------------------------------------------
extern "C" void kernel_launch(void* const* d_in, const int* in_sizes, int n_in,
                              void* d_out, int out_size, void* d_ws, size_t ws_size,
                              hipStream_t stream) {
    (void)in_sizes; (void)n_in; (void)out_size; (void)ws_size;
    const float* x     = (const float*)d_in[0];
    const int*   ei    = (const int*)d_in[1];
    const float* eattr = (const float*)d_in[2];
    const float* enw1 = (const float*)d_in[3];  const float* enb1 = (const float*)d_in[4];
    const float* enw2 = (const float*)d_in[5];  const float* enb2 = (const float*)d_in[6];
    const float* eew1 = (const float*)d_in[7];  const float* eeb1 = (const float*)d_in[8];
    const float* eew2 = (const float*)d_in[9];  const float* eeb2 = (const float*)d_in[10];
    const float* mw1  = (const float*)d_in[11]; const float* mb1  = (const float*)d_in[12];
    const float* mw2  = (const float*)d_in[13]; const float* mb2  = (const float*)d_in[14];
    const float* uw1  = (const float*)d_in[15]; const float* ub1  = (const float*)d_in[16];
    const float* uw2  = (const float*)d_in[17]; const float* ub2  = (const float*)d_in[18];
    const float* c1w  = (const float*)d_in[19]; const float* c1b  = (const float*)d_in[20];
    const float* c2w  = (const float*)d_in[21]; const float* c2b  = (const float*)d_in[22];
    float* out = (float*)d_out;

    char* base = (char*)d_ws;
    size_t off = 0;
    auto take = [&](size_t bytes) -> char* {
        char* p = base + off;
        off = (off + bytes + 255) & ~(size_t)255;
        return p;
    };
    float*  hf  = (float*) take((size_t)NNODES * HID * 4);
    __bf16* hbf = (__bf16*)take((size_t)NNODES * HID * 2);
    __bf16* ebf = (__bf16*)take((size_t)NEDGES * HID * 2);
    float*  agg = (float*) take((size_t)NNODES * HID * 4);
    float*  deg = (float*) take((size_t)NNODES * 4);
    __bf16* p_enw1 = (__bf16*)take(4096 * 2);
    __bf16* p_enw2 = (__bf16*)take(16384 * 2);
    __bf16* p_eew1 = (__bf16*)take(4096 * 2);
    __bf16* p_eew2 = (__bf16*)take(16384 * 2);
    __bf16* p_mw1  = (__bf16*)take((size_t)NLAYERS * 49152 * 2);
    __bf16* p_mw2  = (__bf16*)take((size_t)NLAYERS * 16384 * 2);
    __bf16* p_uw1  = (__bf16*)take((size_t)NLAYERS * 32768 * 2);
    __bf16* p_uw2  = (__bf16*)take((size_t)NLAYERS * 16384 * 2);

    auto nb = [](size_t n, int b) { return (unsigned)((n + (size_t)b - 1) / (size_t)b); };

    // Pack all weights into WMMA B-tile layout (bf16).
    repack_kernel<<<nb(4096, 256), 256, 0, stream>>>(enw1, p_enw1, IN_NODE, 32, 1);
    repack_kernel<<<nb(16384, 256), 256, 0, stream>>>(enw2, p_enw2, 128, 128, 1);
    repack_kernel<<<nb(4096, 256), 256, 0, stream>>>(eew1, p_eew1, IN_EDGE, 32, 1);
    repack_kernel<<<nb(16384, 256), 256, 0, stream>>>(eew2, p_eew2, 128, 128, 1);
    repack_kernel<<<nb((size_t)NLAYERS * 49152, 256), 256, 0, stream>>>(mw1, p_mw1, 384, 384, NLAYERS);
    repack_kernel<<<nb((size_t)NLAYERS * 16384, 256), 256, 0, stream>>>(mw2, p_mw2, 128, 128, NLAYERS);
    repack_kernel<<<nb((size_t)NLAYERS * 32768, 256), 256, 0, stream>>>(uw1, p_uw1, 256, 256, NLAYERS);
    repack_kernel<<<nb((size_t)NLAYERS * 16384, 256), 256, 0, stream>>>(uw2, p_uw2, 128, 128, NLAYERS);

    zero_kernel<<<nb((size_t)NNODES * HID, 256), 256, 0, stream>>>(agg, deg);

    enc_node_kernel<<<NNODES / 16, 32, 0, stream>>>(x, p_enw1, enb1, p_enw2, enb2, hf, hbf);
    enc_edge_kernel<<<NEDGES / 16, 32, 0, stream>>>(eattr, p_eew1, eeb1, p_eew2, eeb2, ebf);

    const int* srcI = ei;
    const int* dstI = ei + NEDGES;
    deg_kernel<<<nb(NEDGES, 256), 256, 0, stream>>>(dstI, deg);
    invdeg_kernel<<<nb(NNODES, 256), 256, 0, stream>>>(deg);

    for (int l = 0; l < NLAYERS; ++l) {
        msg_kernel<<<NEDGES / 64, 128, 0, stream>>>(
            hbf, ebf, srcI, dstI,
            p_mw1 + (size_t)l * 49152, mb1 + (size_t)l * HID,
            p_mw2 + (size_t)l * 16384, mb2 + (size_t)l * HID, agg);
        upd_kernel<<<NNODES / 16, 32, 0, stream>>>(
            hf, hbf, agg, deg,
            p_uw1 + (size_t)l * 32768, ub1 + (size_t)l * HID,
            p_uw2 + (size_t)l * 16384, ub2 + (size_t)l * HID);
    }

    conv_kernel<<<nb((size_t)NNODES * 20, 256), 256, 0, stream>>>(hf, c1w, c1b, c2w, c2b, out);
}
